// GraphEncoderBlockSE_36910948941898
// MI455X (gfx1250) — matrix operand order repaired and checked by reading, not compiled, not run on hardware
//
#include <hip/hip_runtime.h>
#include <hip/hip_bf16.h>
#include <cstdint>

// ---- problem constants (match reference) ----
#define N_NODES 512
#define IMG2    1024                 // 32*32
#define FDIM    3072                 // C*IMG*IMG
#define HEADS   4
#define NH_F    12288                // HEADS*F
#define NTOT    24576                // 2*HEADS*F  (xl | xr)
#define E_EDGES 4096
#define E_TOT   (E_EDGES + N_NODES)  // + self loops = 4608
#define EPSB    1e-5f
#define SLOPE   0.2f

// TDM staging geometry for the GEMM A panel
#define KHALF     1536               // K elements staged per TDM op
#define PAD_BYTES 176                // 44 dwords inserted every 1024B of stream
#define ROW_PITCH 3600               // 1536*2 + 3*176  (==900 dw, 900%64==4 -> conflict-free)
#define LDS_BYTES (16 * ROW_PITCH)   // 57600 B

typedef __bf16 bf16_t;
typedef __attribute__((ext_vector_type(16))) __bf16 v16bf;
typedef __attribute__((ext_vector_type(8)))  float  v8f;
typedef __attribute__((ext_vector_type(4)))  unsigned int u32x4;
typedef __attribute__((ext_vector_type(8)))  int    i32x8;
typedef __attribute__((ext_vector_type(4)))  int    i32x4;
typedef __attribute__((ext_vector_type(4)))  float  f32x4;

// ------------------------------------------------------------------
// 1) fc1 (1x1-conv) + BN -> hf in bf16  [512, 3072]
// ------------------------------------------------------------------
__global__ void fc1_bn_kernel(const float* __restrict__ x,
                              const float* __restrict__ W,  const float* __restrict__ b,
                              const float* __restrict__ g,  const float* __restrict__ bb,
                              const float* __restrict__ m,  const float* __restrict__ v,
                              bf16_t* __restrict__ hf) {
  int gid = blockIdx.x * blockDim.x + threadIdx.x;       // over N*1024 pixels
  if (gid >= N_NODES * IMG2) return;
  int n = gid >> 10, hw = gid & 1023;
  const float* xp = x + (size_t)n * FDIM + hw;
  float x0 = xp[0], x1 = xp[1024], x2 = xp[2048];
  bf16_t* out = hf + (size_t)n * FDIM + hw;
#pragma unroll
  for (int c = 0; c < 3; ++c) {
    float acc = W[c*3+0]*x0 + W[c*3+1]*x1 + W[c*3+2]*x2 + b[c];
    float inv = g[c] * rsqrtf(v[c] + EPSB);
    acc = acc * inv + (bb[c] - m[c] * inv);
    out[c * 1024] = (bf16_t)acc;
  }
}

// ------------------------------------------------------------------
// 2) [Wl | Wr] fp32 [3072, 12288] each  ->  Bt bf16 [24576, 3072]
// ------------------------------------------------------------------
__global__ void wt_transpose_kernel(const float* __restrict__ Wl,
                                    const float* __restrict__ Wr,
                                    bf16_t* __restrict__ Bt) {
  __shared__ float tile[32][33];
  int n0 = blockIdx.x * 32;                // output-column tile (0..24575)
  int k0 = blockIdx.y * 32;                // K tile (0..3071)
  int tx = threadIdx.x & 31, ty = threadIdx.x >> 5;  // 32 x 8
  const float* Wp = (n0 < NH_F) ? (Wl + n0) : (Wr + (n0 - NH_F));
#pragma unroll
  for (int i = 0; i < 4; ++i) {
    int k = ty + 8 * i;
    tile[k][tx] = Wp[(size_t)(k0 + k) * NH_F + tx];
  }
  __syncthreads();
#pragma unroll
  for (int i = 0; i < 4; ++i) {
    int r = ty + 8 * i;                    // output column within tile
    Bt[(size_t)(n0 + r) * FDIM + (k0 + tx)] = (bf16_t)tile[tx][r];
  }
}

// ------------------------------------------------------------------
// 3) WMMA GEMM:  Cm[512, 24576] = A[512,3072]bf16 x Bt^T  (+ bias)
//    * A panel (16 x 3072) staged into LDS by the Tensor Data Mover
//      in two 1536-wide halves with TDM padding for bank-conflict-free
//      ds_load_b128 fragment reads.
//    * B fragments software-pipelined from global (L2-resident bf16).
// ------------------------------------------------------------------
__global__ void __launch_bounds__(256, 1)
gat_gemm_wmma(const bf16_t* __restrict__ A,    // [512][3072]
              const bf16_t* __restrict__ Bt,   // [24576][3072]
              const float* __restrict__ bl,    // [12288]
              const float* __restrict__ br,    // [12288]
              float* __restrict__ Cm) {        // [512][24576]
  __shared__ __align__(128) unsigned char sA[LDS_BYTES];

  const int lane  = threadIdx.x & 31;
  const int wave  = threadIdx.x >> 5;
  const int mBase = blockIdx.y * 16;
  const int nBase = blockIdx.x * 512 + wave * 64;

  // B fragment addressing (contiguous K order k = 16h + s, same as A): 16 contiguous K per lane
  const int ksB = (lane < 16) ? 0 : 16;
  const bf16_t* bPtr = Bt + (size_t)(nBase + (lane & 15)) * FDIM + ksB;

  // A fragment addressing (contiguous K order k = 16h + s, same as B) within padded LDS panel
  const unsigned char* sRow = sA + (lane & 15) * ROW_PITCH;
  const unsigned int kbA2 = ((lane < 16) ? 0 : 16) * 2;   // byte offset of chunk 0

  v8f acc[4];
#pragma unroll
  for (int j = 0; j < 4; ++j) acc[j] = (v8f){0.f,0.f,0.f,0.f,0.f,0.f,0.f,0.f};

  union UV { v16bf v; u32x4 q[2]; };

  for (int half = 0; half < 2; ++half) {
    const int kh = half * KHALF;

    if (wave == 0) {
      // ---- Tensor Data Mover: A[mBase..mBase+15][kh..kh+1535] -> LDS ----
      uint64_t ga = (uint64_t)(uintptr_t)(A + (size_t)mBase * FDIM + kh);
      uint32_t lb = (uint32_t)(uintptr_t)(&sA[0]);       // LDS byte offset
      u32x4 g0;
      g0[0] = 1u;                                        // count=1 (valid user D#)
      g0[1] = lb;                                        // lds_addr
      g0[2] = (uint32_t)ga;                              // global_addr[31:0]
      g0[3] = (uint32_t)((ga >> 32) & 0x01FFFFFFu) | (2u << 30);  // addr hi | type=2
      i32x8 g1;
      g1[0] = (1 << 16)          // data_size = 2 bytes
            | (1 << 20)          // pad_enable
            | (7 << 22)          // pad_interval: 256 dwords (1024 B)
            | (43 << 25);        // pad_amount: 44 dwords (176 B)
      g1[1] = (KHALF & 0xFFFF) << 16;                    // tensor_dim0 lo16
      g1[2] = (KHALF >> 16) | (16 << 16);                // tensor_dim0 hi | tensor_dim1 lo
      g1[3] = (KHALF << 16);                             // tensor_dim1 hi | tile_dim0
      g1[4] = 16;                                        // tile_dim1=16, tile_dim2=0
      g1[5] = FDIM;                                      // tensor_dim0_stride lo32 = 3072
      g1[6] = 0;                                         // stride hi | dim1_stride lo
      g1[7] = 0;
      i32x4 gz = (i32x4){0, 0, 0, 0};
#if __clang_major__ >= 23
      i32x8 gz8 = (i32x8){0, 0, 0, 0, 0, 0, 0, 0};
      __builtin_amdgcn_tensor_load_to_lds(g0, g1, gz, gz, gz8, 0);
#else
      __builtin_amdgcn_tensor_load_to_lds(g0, g1, gz, gz, 0);
#endif
      __builtin_amdgcn_s_wait_tensorcnt(0);
    }
    __syncthreads();   // TDM writes visible to all waves

    // ---- software-pipelined K loop over this half ----
    UV bcur[4];
    {
      const bf16_t* bp0 = bPtr + kh;
#pragma unroll
      for (int j = 0; j < 4; ++j) {
        const bf16_t* bp = bp0 + (size_t)j * 16 * FDIM;
        bcur[j].q[0] = *(const u32x4*)bp;
        bcur[j].q[1] = *(const u32x4*)(bp + 8);
      }
    }

    for (int k0 = 0; k0 < KHALF; k0 += 32) {
      // prefetch next-step B fragments into independent registers
      UV bnxt[4];
      const int kn = kh + ((k0 + 32 < KHALF) ? (k0 + 32) : k0);  // clamp at tail
      const bf16_t* bpn = bPtr + kn;
#pragma unroll
      for (int j = 0; j < 4; ++j) {
        const bf16_t* bp = bpn + (size_t)j * 16 * FDIM;
        bnxt[j].q[0] = *(const u32x4*)bp;
        bnxt[j].q[1] = *(const u32x4*)(bp + 8);
      }
      // A fragment from padded LDS panel (conflict-free ds_load_b128 x2)
      unsigned int e1 = (unsigned int)(k0 * 2) + kbA2;
      unsigned int o1 = e1 + (e1 >> 10) * PAD_BYTES;
      unsigned int e2 = e1 + 16;
      unsigned int o2 = e2 + (e2 >> 10) * PAD_BYTES;
      UV a;
      a.q[0] = *(const u32x4*)(sRow + o1);
      a.q[1] = *(const u32x4*)(sRow + o2);
#pragma unroll
      for (int j = 0; j < 4; ++j)
        acc[j] = __builtin_amdgcn_wmma_f32_16x16x32_bf16(
            false, a.v, false, bcur[j].v, (short)0, acc[j], false, false);
#pragma unroll
      for (int j = 0; j < 4; ++j) bcur[j] = bnxt[j];
    }
    __syncthreads();   // all waves done reading LDS before next TDM overwrite
  }

  // Epilogue: C/D layout -> VGPR i holds M = i (lanes 0-15) / 8+i (lanes 16-31)
  const int rowOff = (lane < 16) ? 0 : 8;
  const int colL   = lane & 15;
#pragma unroll
  for (int j = 0; j < 4; ++j) {
    int col = nBase + j * 16 + colL;
    float bias = (col < NH_F) ? bl[col] : br[col - NH_F];
    float* cp = Cm + (size_t)(mBase + rowOff) * NTOT + col;
#pragma unroll
    for (int i = 0; i < 8; ++i)
      cp[(size_t)i * NTOT] = acc[j][i] + bias;
  }
}

// ------------------------------------------------------------------
// 4) GATv2 edge logits: one block per edge, one wave per head
// ------------------------------------------------------------------
__global__ void edge_logits_kernel(const int* __restrict__ edge_index,
                                   const float* __restrict__ xlr,
                                   const float* __restrict__ att,
                                   float* __restrict__ logits) {
  int e = blockIdx.x;
  int h = threadIdx.x >> 5;
  int lane = threadIdx.x & 31;
  int s, d;
  if (e < E_EDGES) { s = edge_index[e]; d = edge_index[E_EDGES + e]; }
  else             { s = d = e - E_EDGES; }
  const float* xl = xlr + (size_t)s * NTOT + h * FDIM;
  const float* xr = xlr + (size_t)d * NTOT + NH_F + h * FDIM;
  const float* ap = att + h * FDIM;
  float acc = 0.f;
  for (int f = lane; f < FDIM; f += 32) {
    float val = xl[f] + xr[f];
    val = (val > 0.f) ? val : SLOPE * val;
    acc += ap[f] * val;
  }
#pragma unroll
  for (int off = 16; off > 0; off >>= 1)
    acc += __shfl_xor(acc, off, 32);
  if (lane == 0) logits[e * HEADS + h] = acc;
}

// ------------------------------------------------------------------
// 5) per-destination segment max & exp-sum (deterministic, no atomics)
// ------------------------------------------------------------------
__global__ void segment_stats_kernel(const int* __restrict__ edge_index,
                                     const float* __restrict__ logits,
                                     float* __restrict__ maxv,
                                     float* __restrict__ denom) {
  int n = blockIdx.x;
  int tid = threadIdx.x;               // 128 threads
  __shared__ float red[128][4];
  float mx[4] = {-1e30f, -1e30f, -1e30f, -1e30f};
  for (int e = tid; e < E_TOT; e += 128) {
    int d = (e < E_EDGES) ? edge_index[E_EDGES + e] : (e - E_EDGES);
    if (d != n) continue;
#pragma unroll
    for (int h = 0; h < 4; ++h) mx[h] = fmaxf(mx[h], logits[e * 4 + h]);
  }
#pragma unroll
  for (int h = 0; h < 4; ++h) red[tid][h] = mx[h];
  __syncthreads();
  for (int s = 64; s > 0; s >>= 1) {
    if (tid < s) {
#pragma unroll
      for (int h = 0; h < 4; ++h)
        red[tid][h] = fmaxf(red[tid][h], red[tid + s][h]);
    }
    __syncthreads();
  }
  float m0[4];
#pragma unroll
  for (int h = 0; h < 4; ++h) m0[h] = red[0][h];
  __syncthreads();

  float sm[4] = {0.f, 0.f, 0.f, 0.f};
  for (int e = tid; e < E_TOT; e += 128) {
    int d = (e < E_EDGES) ? edge_index[E_EDGES + e] : (e - E_EDGES);
    if (d != n) continue;
#pragma unroll
    for (int h = 0; h < 4; ++h) sm[h] += __expf(logits[e * 4 + h] - m0[h]);
  }
#pragma unroll
  for (int h = 0; h < 4; ++h) red[tid][h] = sm[h];
  __syncthreads();
  for (int s = 64; s > 0; s >>= 1) {
    if (tid < s) {
#pragma unroll
      for (int h = 0; h < 4; ++h) red[tid][h] += red[tid + s][h];
    }
    __syncthreads();
  }
  if (tid == 0) {
#pragma unroll
    for (int h = 0; h < 4; ++h) {
      maxv[n * 4 + h]  = m0[h];
      denom[n * 4 + h] = red[0][h];
    }
  }
}

// ------------------------------------------------------------------
// 6) weighted aggregation: one block per node, 48 floats per thread
// ------------------------------------------------------------------
__global__ void aggregate_kernel(const int* __restrict__ edge_index,
                                 const float* __restrict__ xlr,
                                 const float* __restrict__ logits,
                                 const float* __restrict__ maxv,
                                 const float* __restrict__ denom,
                                 const float* __restrict__ gat_b,
                                 float* __restrict__ g_out) {
  int n = blockIdx.x;
  int tid = threadIdx.x;               // 256 threads
  int head = tid / 64;                 // (tid*48)/3072
  int p0 = tid * 48;
  float acc[48];
#pragma unroll
  for (int k = 0; k < 48; ++k) acc[k] = 0.f;
  float mh = maxv[n * 4 + head];
  float dh = denom[n * 4 + head];
  for (int e = 0; e < E_TOT; ++e) {
    int s, d;
    if (e < E_EDGES) { s = edge_index[e]; d = edge_index[E_EDGES + e]; }
    else             { s = d = e - E_EDGES; }
    if (d != n) continue;
    float alpha = __expf(logits[e * 4 + head] - mh) / dh;
    const float* xp = xlr + (size_t)s * NTOT + p0;   // xl half
#pragma unroll
    for (int k = 0; k < 48; k += 4) {
      f32x4 v4 = *(const f32x4*)(xp + k);
      acc[k+0] += alpha * v4.x;  acc[k+1] += alpha * v4.y;
      acc[k+2] += alpha * v4.z;  acc[k+3] += alpha * v4.w;
    }
  }
  float* op = g_out + (size_t)n * NH_F + p0;
  const float* bp = gat_b + p0;
#pragma unroll
  for (int k = 0; k < 48; ++k) op[k] = acc[k] + bp[k];
}

// ------------------------------------------------------------------
// 7) tail: fc2+BN2+shortcut, FFN(relu-bn conv) x2 with residual
// ------------------------------------------------------------------
__global__ void tail_kernel(const float* __restrict__ x,
                            const float* __restrict__ g_out,
                            const float* __restrict__ fc2_W, const float* __restrict__ fc2_b,
                            const float* __restrict__ bn2_g, const float* __restrict__ bn2_b,
                            const float* __restrict__ bn2_m, const float* __restrict__ bn2_v,
                            const float* __restrict__ ffn1_W, const float* __restrict__ ffn1_b,
                            const float* __restrict__ bnf1_g, const float* __restrict__ bnf1_b,
                            const float* __restrict__ bnf1_m, const float* __restrict__ bnf1_v,
                            const float* __restrict__ ffn2_W, const float* __restrict__ ffn2_b,
                            const float* __restrict__ bnf2_g, const float* __restrict__ bnf2_b,
                            const float* __restrict__ bnf2_m, const float* __restrict__ bnf2_v,
                            float* __restrict__ out) {
  int gid = blockIdx.x * blockDim.x + threadIdx.x;   // N*1024 pixels
  if (gid >= N_NODES * IMG2) return;
  int n = gid >> 10, hw = gid & 1023;

  float g12[12];
#pragma unroll
  for (int j = 0; j < 12; ++j)
    g12[j] = g_out[(size_t)n * NH_F + j * 1024 + hw];

  float gch[3];
#pragma unroll
  for (int c = 0; c < 3; ++c) {
    float acc = fc2_b[c];
#pragma unroll
    for (int j = 0; j < 12; ++j) acc += fc2_W[c * 12 + j] * g12[j];
    float inv = bn2_g[c] * rsqrtf(bn2_v[c] + EPSB);
    acc = acc * inv + (bn2_b[c] - bn2_m[c] * inv);
    gch[c] = acc + x[(size_t)n * FDIM + c * 1024 + hw];   // shortcut
  }

  float f1[3];
#pragma unroll
  for (int c = 0; c < 3; ++c) {
    float acc = ffn1_b[c];
#pragma unroll
    for (int j = 0; j < 3; ++j) acc += ffn1_W[c * 3 + j] * gch[j];
    float inv = bnf1_g[c] * rsqrtf(bnf1_v[c] + EPSB);
    acc = acc * inv + (bnf1_b[c] - bnf1_m[c] * inv);
    f1[c] = fmaxf(acc, 0.f);
  }

#pragma unroll
  for (int c = 0; c < 3; ++c) {
    float acc = ffn2_b[c];
#pragma unroll
    for (int j = 0; j < 3; ++j) acc += ffn2_W[c * 3 + j] * f1[j];
    float inv = bnf2_g[c] * rsqrtf(bnf2_v[c] + EPSB);
    acc = acc * inv + (bnf2_b[c] - bnf2_m[c] * inv);
    out[(size_t)n * FDIM + c * 1024 + hw] = acc + gch[c];
  }
}

// ------------------------------------------------------------------
extern "C" void kernel_launch(void* const* d_in, const int* in_sizes, int n_in,
                              void* d_out, int out_size, void* d_ws, size_t ws_size,
                              hipStream_t stream) {
  (void)in_sizes; (void)n_in; (void)out_size; (void)ws_size;
  const float* x         = (const float*)d_in[0];
  const int*   edge_index= (const int*)  d_in[1];
  const float* fc1_W = (const float*)d_in[2];
  const float* fc1_b = (const float*)d_in[3];
  const float* bn1_g = (const float*)d_in[4];
  const float* bn1_b = (const float*)d_in[5];
  const float* bn1_m = (const float*)d_in[6];
  const float* bn1_v = (const float*)d_in[7];
  const float* Wl    = (const float*)d_in[8];
  const float* bl    = (const float*)d_in[9];
  const float* Wr    = (const float*)d_in[10];
  const float* br    = (const float*)d_in[11];
  const float* att   = (const float*)d_in[12];
  const float* gat_b = (const float*)d_in[13];
  const float* fc2_W = (const float*)d_in[14];
  const float* fc2_b = (const float*)d_in[15];
  const float* bn2_g = (const float*)d_in[16];
  const float* bn2_b = (const float*)d_in[17];
  const float* bn2_m = (const float*)d_in[18];
  const float* bn2_v = (const float*)d_in[19];
  const float* ffn1_W = (const float*)d_in[20];
  const float* ffn1_b = (const float*)d_in[21];
  const float* bnf1_g = (const float*)d_in[22];
  const float* bnf1_b = (const float*)d_in[23];
  const float* bnf1_m = (const float*)d_in[24];
  const float* bnf1_v = (const float*)d_in[25];
  const float* ffn2_W = (const float*)d_in[26];
  const float* ffn2_b = (const float*)d_in[27];
  const float* bnf2_g = (const float*)d_in[28];
  const float* bnf2_b = (const float*)d_in[29];
  const float* bnf2_m = (const float*)d_in[30];
  const float* bnf2_v = (const float*)d_in[31];

  // workspace carve-out (256B aligned slices)
  char* ws = (char*)d_ws;
  size_t off = 0;
  auto carve = [&](size_t bytes) -> void* {
    void* p = ws + off;
    off += (bytes + 255) & ~(size_t)255;
    return p;
  };
  bf16_t* hf     = (bf16_t*)carve((size_t)N_NODES * FDIM * sizeof(bf16_t));   // 3 MB
  bf16_t* Bt     = (bf16_t*)carve((size_t)NTOT * FDIM * sizeof(bf16_t));      // 151 MB
  float*  xlr    = (float*) carve((size_t)N_NODES * NTOT * sizeof(float));    // 50 MB
  float*  logits = (float*) carve((size_t)E_TOT * HEADS * sizeof(float));
  float*  maxv   = (float*) carve((size_t)N_NODES * HEADS * sizeof(float));
  float*  denom  = (float*) carve((size_t)N_NODES * HEADS * sizeof(float));
  float*  g_out  = (float*) carve((size_t)N_NODES * NH_F * sizeof(float));    // 25 MB

  // 1) fc1 + BN -> bf16 node features
  fc1_bn_kernel<<<(N_NODES * IMG2 + 255) / 256, 256, 0, stream>>>(
      x, fc1_W, fc1_b, bn1_g, bn1_b, bn1_m, bn1_v, hf);

  // 2) transpose + convert weights -> Bt (L2-resident bf16)
  wt_transpose_kernel<<<dim3(NTOT / 32, FDIM / 32), 256, 0, stream>>>(Wl, Wr, Bt);

  // 3) big WMMA GEMM (TDM-staged A panel): xl|xr = hf x [Wl|Wr] + bias
  gat_gemm_wmma<<<dim3(NTOT / 512, N_NODES / 16), 256, 0, stream>>>(hf, Bt, bl, br, xlr);

  // 4) edge logits
  edge_logits_kernel<<<E_TOT, 128, 0, stream>>>(edge_index, xlr, att, logits);

  // 5) segment softmax stats
  segment_stats_kernel<<<N_NODES, 128, 0, stream>>>(edge_index, logits, maxv, denom);

  // 6) aggregation (+ gat bias)
  aggregate_kernel<<<N_NODES, 256, 0, stream>>>(edge_index, xlr, logits, maxv, denom,
                                                gat_b, g_out);

  // 7) fc2+BN+shortcut, FFN, residual -> out
  tail_kernel<<<(N_NODES * IMG2 + 255) / 256, 256, 0, stream>>>(
      x, g_out, fc2_W, fc2_b, bn2_g, bn2_b, bn2_m, bn2_v,
      ffn1_W, ffn1_b, bnf1_g, bnf1_b, bnf1_m, bnf1_v,
      ffn2_W, ffn2_b, bnf2_g, bnf2_b, bnf2_m, bnf2_v,
      (float*)d_out);
}
